// ClusterOfExpertsLayer_54288386622045
// MI455X (gfx1250) — compile-verified, matrix operand order checked by reading, not run on hardware
//
#include <hip/hip_runtime.h>
#include <hip/hip_bf16.h>
#include <math.h>

// ---------------------------------------------------------------------------
// ClusterOfExperts layer for MI455X (gfx1250), wave32 + v_wmma_f32_16x16x32_f16
// Double-buffered (LDS ping-pong + register prefetch) WMMA GEMM pipeline.
// ---------------------------------------------------------------------------

#define TOK 2048   // B*S
#define DIM 1024   // D
#define FF  2048   // F
#define NC  4      // clusters
#define NE  4      // experts per cluster

typedef __attribute__((ext_vector_type(16))) _Float16 v16h;
typedef __attribute__((ext_vector_type(8)))  _Float16 v8h;
typedef __attribute__((ext_vector_type(8)))  float    v8f;

// -------------------------- utility kernels --------------------------------

__global__ __launch_bounds__(256) void zero_f32(float* __restrict__ p, size_t n) {
  size_t i = (size_t)blockIdx.x * 256 + threadIdx.x;
  size_t stride = (size_t)gridDim.x * 256;
  for (; i < n; i += stride) p[i] = 0.0f;
}

// -------------------------- router -----------------------------------------
__global__ __launch_bounds__(256) void router_kernel(
    const float* __restrict__ x,    // [T,D]
    const float* __restrict__ Wrc,  // [D,C]
    const float* __restrict__ brc,  // [C]
    const float* __restrict__ Wre,  // [C,D,E]
    const float* __restrict__ bre,  // [C,E]
    float* __restrict__ gate,       // [T,C]
    float* __restrict__ coef)       // [T,C,E]
{
  const int t = blockIdx.x;
  const int tid = threadIdx.x;
  __shared__ float red[256];
  __shared__ float logits[20];

  float p[20];
#pragma unroll
  for (int o = 0; o < 20; ++o) p[o] = 0.0f;

  for (int d = tid; d < DIM; d += 256) {
    float xv = x[(size_t)t * DIM + d];
#pragma unroll
    for (int c = 0; c < NC; ++c) p[c] += xv * Wrc[(size_t)d * NC + c];
#pragma unroll
    for (int c = 0; c < NC; ++c)
#pragma unroll
      for (int e = 0; e < NE; ++e)
        p[4 + c * NE + e] += xv * Wre[((size_t)c * DIM + d) * NE + e];
  }

  for (int o = 0; o < 20; ++o) {
    red[tid] = p[o];
    __syncthreads();
    for (int s = 128; s > 0; s >>= 1) {
      if (tid < s) red[tid] += red[tid + s];
      __syncthreads();
    }
    if (tid == 0) logits[o] = red[0];
    __syncthreads();
  }

  if (tid == 0) {
    // ----- level 1: cluster softmax -> top-2 -> renormalize -----
    float cl[NC];
#pragma unroll
    for (int c = 0; c < NC; ++c) cl[c] = logits[c] + brc[c];
    float m = cl[0];
#pragma unroll
    for (int c = 1; c < NC; ++c) m = fmaxf(m, cl[c]);
    float ex[NC], s = 0.0f;
#pragma unroll
    for (int c = 0; c < NC; ++c) { ex[c] = expf(cl[c] - m); s += ex[c]; }
    float pr[NC];
#pragma unroll
    for (int c = 0; c < NC; ++c) pr[c] = ex[c] / s;
    int i0 = 0;
    for (int c = 1; c < NC; ++c) if (pr[c] > pr[i0]) i0 = c;
    int i1 = -1;
    for (int c = 0; c < NC; ++c) {
      if (c == i0) continue;
      if (i1 < 0 || pr[c] > pr[i1]) i1 = c;
    }
    float s2 = pr[i0] + pr[i1];
    float g[NC] = {0.f, 0.f, 0.f, 0.f};
    g[i0] = pr[i0] / s2;
    g[i1] = pr[i1] / s2;
#pragma unroll
    for (int c = 0; c < NC; ++c) gate[(size_t)t * NC + c] = g[c];

    // ----- level 2: per-cluster expert top-2 + softmax -----
    for (int c = 0; c < NC; ++c) {
      float el[NE];
#pragma unroll
      for (int e = 0; e < NE; ++e) el[e] = logits[4 + c * NE + e] + bre[c * NE + e];
      int j0 = 0;
      for (int e = 1; e < NE; ++e) if (el[e] > el[j0]) j0 = e;
      int j1 = -1;
      for (int e = 0; e < NE; ++e) {
        if (e == j0) continue;
        if (j1 < 0 || el[e] > el[j1]) j1 = e;
      }
      float mm = fmaxf(el[j0], el[j1]);
      float e0 = expf(el[j0] - mm), e1 = expf(el[j1] - mm);
      float ss = e0 + e1;
      float cc[NE] = {0.f, 0.f, 0.f, 0.f};
      cc[j0] = e0 / ss;
      cc[j1] = e1 / ss;
#pragma unroll
      for (int e = 0; e < NE; ++e) coef[((size_t)t * NC + c) * NE + e] = cc[e];
    }
  }
}

// ------------------ row LayerNorm (f32 in -> f16 out) ----------------------
__global__ __launch_bounds__(256) void ln_rows_f16(
    const float* __restrict__ X, const float* __restrict__ g,
    const float* __restrict__ b, _Float16* __restrict__ out, int ncols)
{
  const int t = blockIdx.x, tid = threadIdx.x;
  const float* row = X + (size_t)t * ncols;
  __shared__ float r1[256], r2[256];
  float s = 0.f, s2 = 0.f;
  for (int d = tid; d < ncols; d += 256) {
    float v = row[d];
    s += v; s2 += v * v;
  }
  r1[tid] = s; r2[tid] = s2;
  __syncthreads();
  for (int st = 128; st > 0; st >>= 1) {
    if (tid < st) { r1[tid] += r1[tid + st]; r2[tid] += r2[tid + st]; }
    __syncthreads();
  }
  float mean = r1[0] / ncols;
  float var  = r2[0] / ncols - mean * mean;
  float inv  = rsqrtf(var + 1e-5f);
  for (int d = tid; d < ncols; d += 256) {
    float v = (row[d] - mean) * inv * g[d] + b[d];
    out[(size_t)t * ncols + d] = (_Float16)v;
  }
}

// -------- per-expert output LN + coef-weighted accumulate into Oc ----------
__global__ __launch_bounds__(256) void y_ln_accum(
    const float* __restrict__ Y, const float* __restrict__ g,
    const float* __restrict__ b, const float* __restrict__ coef,
    int c, int e, float* __restrict__ Oc)
{
  const int t = blockIdx.x, tid = threadIdx.x;
  const float cf = coef[((size_t)t * NC + c) * NE + e];
  if (cf == 0.0f) return;  // uniform per block: top-2 sparsity
  const float* row = Y + (size_t)t * DIM;
  __shared__ float r1[256], r2[256];
  float s = 0.f, s2 = 0.f;
  for (int d = tid; d < DIM; d += 256) {
    float v = row[d];
    s += v; s2 += v * v;
  }
  r1[tid] = s; r2[tid] = s2;
  __syncthreads();
  for (int st = 128; st > 0; st >>= 1) {
    if (tid < st) { r1[tid] += r1[tid + st]; r2[tid] += r2[tid + st]; }
    __syncthreads();
  }
  float mean = r1[0] / DIM;
  float var  = r2[0] / DIM - mean * mean;
  float inv  = rsqrtf(var + 1e-5f);
  for (int d = tid; d < DIM; d += 256) {
    float v = (row[d] - mean) * inv * g[d] + b[d];
    Oc[(size_t)t * DIM + d] += cf * v;
  }
}

// -------- per-cluster LN + gate-weighted accumulate into out ---------------
__global__ __launch_bounds__(256) void cluster_finish(
    const float* __restrict__ Oc, const float* __restrict__ g,
    const float* __restrict__ b, const float* __restrict__ gate,
    int c, float* __restrict__ out)
{
  const int t = blockIdx.x, tid = threadIdx.x;
  const float gc = gate[(size_t)t * NC + c];
  if (gc == 0.0f) return;  // uniform per block: top-2 cluster sparsity
  const float* row = Oc + (size_t)t * DIM;
  __shared__ float r1[256], r2[256];
  float s = 0.f, s2 = 0.f;
  for (int d = tid; d < DIM; d += 256) {
    float v = row[d];
    s += v; s2 += v * v;
  }
  r1[tid] = s; r2[tid] = s2;
  __syncthreads();
  for (int st = 128; st > 0; st >>= 1) {
    if (tid < st) { r1[tid] += r1[tid + st]; r2[tid] += r2[tid + st]; }
    __syncthreads();
  }
  float mean = r1[0] / DIM;
  float var  = r2[0] / DIM - mean * mean;
  float inv  = rsqrtf(var + 1e-5f);
  for (int d = tid; d < DIM; d += 256) {
    float v = (row[d] - mean) * inv * g[d] + b[d];
    out[(size_t)t * DIM + d] += gc * v;
  }
}

// --------------------------- WMMA GEMM -------------------------------------
// C[M,N] = A[M,K](f16) @ B[K,N](f32, converted to f16 in flight)
// 256 threads = 8 waves; block tile 128x128x64; wave tile 32x64 = 2x4 WMMAs
// per 32-deep k-slice (16 WMMAs per barrier). LDS ping-pong + register
// prefetch: next tile's global loads are issued before the current tile's
// WMMAs so HBM/L2 latency hides behind the matrix pipe.
// Epilogue modes:
//   0: Cf32 = acc
//   1: Cf32 = acc + aux            (residual add, aux = x)
//   2: Ch16 = gelu_exact(acc)
//   3: Ch16 = relu(acc)
//   4: Ch16 = silu(acc)*acc*aux    (swiglu combine; aux = h1 f32)
#define BM 128
#define BN 128
#define BK 64
#define LSTR 72   // padded LDS row stride in f16 (144B, 16B-aligned)

__global__ __launch_bounds__(256) void gemm_wmma_f16(
    const _Float16* __restrict__ A,  // [M,K] f16 row-major
    const float* __restrict__ B,     // [K,N] f32 row-major
    const float* __restrict__ aux,   // epilogue auxiliary (may be null)
    float* __restrict__ Cf32,        // f32 output (modes 0,1)
    _Float16* __restrict__ Ch16,     // f16 output (modes 2,3,4)
    int M, int N, int K, int mode)
{
  __shared__ _Float16 lA[2][BM * LSTR];
  __shared__ _Float16 lB[2][BN * LSTR];

  const int tid  = threadIdx.x;
  const int bm   = blockIdx.y * BM;
  const int bn   = blockIdx.x * BN;
  const int wave = tid >> 5;
  const int lane = tid & 31;
  const int hlf  = lane >> 4;   // 16-lane half of the wave
  const int l16  = lane & 15;
  const int wm   = (wave & 3) * 32;   // wave row base within block tile
  const int wn   = (wave >> 2) * 64;  // wave col base within block tile

  // staging assignment: 256 threads cover 128 rows/cols x two 32-wide k-segments
  const int ar   = tid & 127;          // A row this thread stages
  const int aseg = (tid >> 7) * 32;    // k-segment base: 0 or 32
  const int bcol = tid & 127;          // B column this thread stages
  const int bseg = (tid >> 7) * 32;

  v8f acc[2][4] = {};

  // ---- prologue: stage tile 0 into buffer 0 ----
  {
    const _Float16* gA = A + (size_t)(bm + ar) * K + aseg;
#pragma unroll
    for (int q = 0; q < 4; ++q)
      *(v8h*)&lA[0][ar * LSTR + aseg + q * 8] = *(const v8h*)(gA + q * 8);
    const float* gB = B + (size_t)bseg * N + bn + bcol;
#pragma unroll
    for (int q = 0; q < 4; ++q) {
      v8h t;
#pragma unroll
      for (int kk = 0; kk < 8; ++kk) t[kk] = (_Float16)gB[(size_t)(q * 8 + kk) * N];
      *(v8h*)&lB[0][bcol * LSTR + bseg + q * 8] = t;
    }
  }
  __syncthreads();

  const int nk = K / BK;
  for (int kt = 0; kt < nk; ++kt) {
    const int cur = kt & 1;
    const bool more = (kt + 1 < nk);

    // ---- phase 1: issue next-tile global loads into registers ----
    v8h   aNext[4];
    float bNext[32];
    if (more) {
      const int k0 = (kt + 1) * BK;
      const _Float16* gA = A + (size_t)(bm + ar) * K + k0 + aseg;
#pragma unroll
      for (int q = 0; q < 4; ++q) aNext[q] = *(const v8h*)(gA + q * 8);
      const float* gB = B + (size_t)(k0 + bseg) * N + bn + bcol;
#pragma unroll
      for (int kk = 0; kk < 32; ++kk) bNext[kk] = gB[(size_t)kk * N];
    }

    // ---- phase 2: compute on current tile (two 32-deep k-slices) ----
#pragma unroll
    for (int ks = 0; ks < 2; ++ks) {
      const int kof = ks * 32;
      v16h afrag[2], bfrag[4];
#pragma unroll
      for (int mt = 0; mt < 2; ++mt) {
        const _Float16* p = &lA[cur][(wm + mt * 16 + l16) * LSTR + kof + hlf * 8];
        v8h lo = *(const v8h*)p;
        v8h hi = *(const v8h*)(p + 16);
#pragma unroll
        for (int i = 0; i < 8; ++i) { afrag[mt][i] = lo[i]; afrag[mt][i + 8] = hi[i]; }
      }
#pragma unroll
      for (int nt = 0; nt < 4; ++nt) {
        const _Float16* p = &lB[cur][(wn + nt * 16 + l16) * LSTR + kof + hlf * 8];
        v8h lo = *(const v8h*)p;
        v8h hi = *(const v8h*)(p + 16);
#pragma unroll
        for (int i = 0; i < 8; ++i) { bfrag[nt][i] = lo[i]; bfrag[nt][i + 8] = hi[i]; }
      }
#pragma unroll
      for (int mt = 0; mt < 2; ++mt)
#pragma unroll
        for (int nt = 0; nt < 4; ++nt)
          acc[mt][nt] = __builtin_amdgcn_wmma_f32_16x16x32_f16(
              false, afrag[mt], false, bfrag[nt], (short)0, acc[mt][nt],
              false, false);
    }

    // ---- phase 3: convert + store next tile into the other buffer ----
    if (more) {
      const int nxt = cur ^ 1;
#pragma unroll
      for (int q = 0; q < 4; ++q)
        *(v8h*)&lA[nxt][ar * LSTR + aseg + q * 8] = aNext[q];
#pragma unroll
      for (int q = 0; q < 4; ++q) {
        v8h t;
#pragma unroll
        for (int kk = 0; kk < 8; ++kk) t[kk] = (_Float16)bNext[q * 8 + kk];
        *(v8h*)&lB[nxt][bcol * LSTR + bseg + q * 8] = t;
      }
    }
    __syncthreads();
  }

  // ---- epilogue ----
  const float inv_sqrt2 = 0.70710678118654752440f;
#pragma unroll
  for (int mt = 0; mt < 2; ++mt) {
#pragma unroll
    for (int nt = 0; nt < 4; ++nt) {
#pragma unroll
      for (int i = 0; i < 8; ++i) {
        const int row = bm + wm + mt * 16 + hlf * 8 + i;
        const int col = bn + wn + nt * 16 + l16;
        const size_t idx = (size_t)row * N + col;
        float v = acc[mt][nt][i];
        if (mode == 0) {
          Cf32[idx] = v;
        } else if (mode == 1) {
          Cf32[idx] = v + aux[idx];
        } else if (mode == 2) {
          float r = 0.5f * v * (1.0f + erff(v * inv_sqrt2));
          Ch16[idx] = (_Float16)r;
        } else if (mode == 3) {
          Ch16[idx] = (_Float16)fmaxf(v, 0.0f);
        } else {
          float a   = aux[idx];                 // h1
          float sig = 1.0f / (1.0f + expf(-v)); // silu(h2) = h2*sig
          Ch16[idx] = (_Float16)(v * sig * v * a);
        }
      }
    }
  }
}

// --------------------------- host orchestration ----------------------------

extern "C" void kernel_launch(void* const* d_in, const int* in_sizes, int n_in,
                              void* d_out, int out_size, void* d_ws, size_t ws_size,
                              hipStream_t stream) {
  const float* x      = (const float*)d_in[0];   // [B,S,D] -> [T,D]
  const float* W1     = (const float*)d_in[1];   // [C,E,D,F]
  const float* W2     = (const float*)d_in[2];   // [C,E,D,F]
  const float* W3     = (const float*)d_in[3];   // [C,E,F,D]
  const float* lin_g  = (const float*)d_in[4];   // [C,E,D]
  const float* lin_b  = (const float*)d_in[5];
  const float* lout_g = (const float*)d_in[6];
  const float* lout_b = (const float*)d_in[7];
  const float* cln_g  = (const float*)d_in[8];   // [C,D]
  const float* cln_b  = (const float*)d_in[9];
  const float* Wrc    = (const float*)d_in[10];  // [D,C]
  const float* brc    = (const float*)d_in[11];  // [C]
  const float* Wre    = (const float*)d_in[12];  // [C,D,E]
  const float* bre    = (const float*)d_in[13];  // [C,E]
  float* out = (float*)d_out;

  // workspace carve-out
  char* ws = (char*)d_ws;
  size_t off = 0;
  auto carve = [&](size_t bytes) -> void* {
    void* p = ws + off;
    off += (bytes + 255) & ~(size_t)255;
    return p;
  };
  _Float16* xn   = (_Float16*)carve((size_t)TOK * DIM * 2); // LN'd input, f16
  float*    h1   = (float*)   carve((size_t)TOK * FF  * 4); // first matmul, f32
  _Float16* hh   = (_Float16*)carve((size_t)TOK * FF  * 2); // activated hidden, f16
  float*    Y    = (float*)   carve((size_t)TOK * DIM * 4); // x + h@W3, f32
  float*    Oc   = (float*)   carve((size_t)TOK * DIM * 4); // cluster accumulator
  float*    gate = (float*)   carve((size_t)TOK * NC  * 4);
  float*    coef = (float*)   carve((size_t)TOK * NC * NE * 4);
  (void)ws_size; (void)in_sizes; (void)n_in; (void)out_size;

  // out = 0
  zero_f32<<<2048, 256, 0, stream>>>(out, (size_t)TOK * DIM);

  // routing
  router_kernel<<<TOK, 256, 0, stream>>>(x, Wrc, brc, Wre, bre, gate, coef);

  const dim3 gemm_blk(256);
  const dim3 grid_DF(FF / BN, TOK / BM);   // [T,F] output
  const dim3 grid_FD(DIM / BN, TOK / BM);  // [T,D] output

  for (int c = 0; c < NC; ++c) {
    zero_f32<<<2048, 256, 0, stream>>>(Oc, (size_t)TOK * DIM);
    for (int j = 0; j < NE; ++j) {
      const int act = j % 3;  // 0: swiglu, 1: gelu, 2: relu
      const size_t we = (size_t)(c * NE + j);

      // xn = LN(x; lin_g, lin_b) -> f16
      ln_rows_f16<<<TOK, 256, 0, stream>>>(
          x, lin_g + we * DIM, lin_b + we * DIM, xn, DIM);

      if (act == 0) {
        // h1 = xn @ W1 (f32)
        gemm_wmma_f16<<<grid_DF, gemm_blk, 0, stream>>>(
            xn, W1 + we * DIM * FF, nullptr, h1, nullptr,
            TOK, FF, DIM, /*mode=*/0);
        // hh = silu(xn@W2) * (xn@W2) * h1 (fused swiglu epilogue)
        gemm_wmma_f16<<<grid_DF, gemm_blk, 0, stream>>>(
            xn, W2 + we * DIM * FF, h1, nullptr, hh,
            TOK, FF, DIM, /*mode=*/4);
      } else if (act == 1) {
        gemm_wmma_f16<<<grid_DF, gemm_blk, 0, stream>>>(
            xn, W1 + we * DIM * FF, nullptr, nullptr, hh,
            TOK, FF, DIM, /*mode=*/2);
      } else {
        gemm_wmma_f16<<<grid_DF, gemm_blk, 0, stream>>>(
            xn, W1 + we * DIM * FF, nullptr, nullptr, hh,
            TOK, FF, DIM, /*mode=*/3);
      }

      // Y = x + hh @ W3 (residual fused)
      gemm_wmma_f16<<<grid_FD, gemm_blk, 0, stream>>>(
          hh, W3 + we * FF * DIM, x, Y, nullptr,
          TOK, DIM, FF, /*mode=*/1);

      // Oc += coef[:,c,j] * LN(Y; lout_g, lout_b)
      y_ln_accum<<<TOK, 256, 0, stream>>>(
          Y, lout_g + we * DIM, lout_b + we * DIM, coef, c, j, Oc);
    }
    // out += gate[:,c] * LN(Oc; cln_g, cln_b)
    cluster_finish<<<TOK, 256, 0, stream>>>(
        Oc, cln_g + (size_t)c * DIM, cln_b + (size_t)c * DIM, gate, c, out);
  }
}
